// FullNetwork_74277164417474
// MI455X (gfx1250) — matrix operand order repaired and checked by reading, not compiled, not run on hardware
//
#include <hip/hip_runtime.h>

// ---------------- static problem configuration (mirrors reference) ----------------
constexpr int kD    = 20;      // risky assets
constexpr int kH    = 64;      // hidden width
constexpr int kB    = 16384;   // Monte-Carlo batch
constexpr int kNN   = 100;     // SDE steps
constexpr int kDiff = 10;      // steps per policy sub-net
constexpr int kTM   = 128;     // batch rows per workgroup (sim kernel)
constexpr int kAS   = 72;      // LDS activation row stride (halves) -> 144B, 16B aligned
constexpr int kWS   = 72;      // LDS weight row stride (halves)

constexpr float kHstep = 0.01f;
constexpr float kR     = 0.02f;
constexpr float kMu    = 0.06f;
constexpr float kSig   = 0.2f;
constexpr float kCurt  = 0.005f;
constexpr float kInvBN = 0.99950037f;   // 1/sqrt(1 + 1e-3)

typedef __attribute__((ext_vector_type(16))) _Float16 v16h;
typedef __attribute__((ext_vector_type(8)))  _Float16 v8h;
typedef __attribute__((ext_vector_type(8)))  float    v8f;

// Build a 16-half WMMA fragment from two 16B LDS loads.
// A-matrix 16x32 f16 layout: lane L holds M=L&15; halves 0..7 = K base..base+7,
// halves 8..15 = K base+16..base+23, base = (L>=16 ? 8 : 0)  -> hiOff = 16 halves.
// B-matrix 32x16 f16 layout: lane L holds N=L&15; halves 0..7 = K base..base+7, halves 8..15 = K base+16..base+23,
// base = (L>=16 ? 8 : 0), same as A                            -> hiOff = 16 halves.
__device__ __forceinline__ v16h ldfrag(const _Float16* p, int hiOff) {
  v8h lo = *(const v8h*)p;
  v8h hi = *(const v8h*)(p + hiOff);
  return __builtin_shufflevector(lo, hi, 0,1,2,3,4,5,6,7,8,9,10,11,12,13,14,15);
}

// One hidden layer: [128,64](f16 in src) @ [64,64] -> relu/BN epilogue -> f16 dst.
// Each wave computes a 32-row strip (2 M-tiles); 2x4x2 = 16 WMMA per wave.
__device__ __forceinline__ void gemm_hidden(const _Float16* src, _Float16* dst,
                                            const _Float16* w,
                                            const float* bias, const float* scale,
                                            const float* shift, int tid) {
  const int lane  = tid & 31;
  const int wave  = tid >> 5;
  const int laneM = lane & 15;
  const int hiH   = lane >> 4;
#pragma unroll
  for (int mt = 0; mt < 2; ++mt) {
    const int rowbase = wave * 32 + mt * 16;
    const _Float16* ap = src + (rowbase + laneM) * kAS + hiH * 8;
    v16h a0 = ldfrag(ap, 16);        // K = 0..31
    v16h a1 = ldfrag(ap + 32, 16);   // K = 32..63
#pragma unroll
    for (int nt = 0; nt < 4; ++nt) {
      const int col = nt * 16 + laneM;
      const _Float16* bp = w + col * kWS + hiH * 8;
      v16h b0 = ldfrag(bp, 16);
      v16h b1 = ldfrag(bp + 32, 16);
      v8f acc = {0.f,0.f,0.f,0.f,0.f,0.f,0.f,0.f};
      acc = __builtin_amdgcn_wmma_f32_16x16x32_f16(false, a0, false, b0, (short)0, acc, false, false);
      acc = __builtin_amdgcn_wmma_f32_16x16x32_f16(false, a1, false, b1, (short)0, acc, false, false);
      const float bi = bias[col], sc = scale[col], sh = shift[col];
      const int rb = rowbase + hiH * 8;   // C/D layout: VGPR r -> row rb+r, col = lane&15
#pragma unroll
      for (int r = 0; r < 8; ++r) {
        float t = fmaxf(acc[r] + bi, 0.f);
        dst[(rb + r) * kAS + col] = (_Float16)(t * sc + sh);
      }
    }
  }
}

// Output layer: [128,64] @ [64,20 (padded to 32)] -> f32 alphaBuf. 8 WMMA/wave.
__device__ __forceinline__ void gemm_out(const _Float16* src, const _Float16* w,
                                         const float* obias, float* outBuf, int tid) {
  const int lane  = tid & 31;
  const int wave  = tid >> 5;
  const int laneM = lane & 15;
  const int hiH   = lane >> 4;
#pragma unroll
  for (int mt = 0; mt < 2; ++mt) {
    const int rowbase = wave * 32 + mt * 16;
    const _Float16* ap = src + (rowbase + laneM) * kAS + hiH * 8;
    v16h a0 = ldfrag(ap, 16);
    v16h a1 = ldfrag(ap + 32, 16);
#pragma unroll
    for (int nt = 0; nt < 2; ++nt) {
      const int col = nt * 16 + laneM;
      const _Float16* bp = w + col * kWS + hiH * 8;
      v16h b0 = ldfrag(bp, 16);
      v16h b1 = ldfrag(bp + 32, 16);
      v8f acc = {0.f,0.f,0.f,0.f,0.f,0.f,0.f,0.f};
      acc = __builtin_amdgcn_wmma_f32_16x16x32_f16(false, a0, false, b0, (short)0, acc, false, false);
      acc = __builtin_amdgcn_wmma_f32_16x16x32_f16(false, a1, false, b1, (short)0, acc, false, false);
      if (col < kD) {
        const float bi = obias[col];
        const int rb = rowbase + hiH * 8;
#pragma unroll
        for (int r = 0; r < 8; ++r)
          outBuf[(rb + r) * kD + col] = acc[r] + bi;
      }
    }
  }
}

// -------------------- Kernel 1: fused SDE + policy MLPs --------------------
__global__ __launch_bounds__(kTM) void sim_kernel(
    const float* __restrict__ x_in, const float* __restrict__ S0,
    const float* __restrict__ dW,
    const float* __restrict__ u0_W, const float* __restrict__ u0_b,
    const float* __restrict__ uW_in, const float* __restrict__ uW_h,
    const float* __restrict__ ub_h, const float* __restrict__ bn_g,
    const float* __restrict__ bn_b, const float* __restrict__ uW_out,
    const float* __restrict__ ub_out,
    float* __restrict__ wsX, float* __restrict__ wsD,
    float* __restrict__ wsS, float* __restrict__ wsQ) {
  __shared__ __attribute__((aligned(16))) _Float16 actA[kTM * kAS];   // 18432 B
  __shared__ __attribute__((aligned(16))) _Float16 actB[kTM * kAS];   // 18432 B
  __shared__ __attribute__((aligned(16))) _Float16 wt[kH * kWS];      //  9216 B
  __shared__ __attribute__((aligned(16))) _Float16 wtO[32 * kWS];     //  4608 B
  __shared__ float alphaBuf[kTM * kD];                                // 10240 B
  __shared__ float pBias[3 * kH], pScale[3 * kH], pShift[3 * kH];
  __shared__ float pWin[kH], pObias[kD];

  const int tid = threadIdx.x;
  const int b   = blockIdx.x * kTM + tid;

  float x = x_in[b];
  float S[kD], SQ[kD], Np[kD], alpha[kD];
  float dacc = 0.f;
#pragma unroll
  for (int d = 0; d < kD; ++d) {
    S[d]     = S0[b * kD + d];
    SQ[d]    = 1.f;
    Np[d]    = 0.f;
    alpha[d] = x * u0_W[d] + u0_b[d];
  }

  const float sqh  = 0.1f;                          // sqrt(h)
  const float rh   = kR * kHstep;
  const float muh  = kMu * kHstep;
  const float dp   = (kMu - 0.5f * kSig * kSig) * kHstep;
  const float dq   = (kR  - 0.5f * kSig * kSig) * kHstep;
  const float ssq  = kSig * sqh;

#pragma unroll 1
  for (int n = 1; n <= kNN; ++n) {
    const float* dwp = dW + ((size_t)(n - 1) * kB + b) * kD;
    float dw[kD];
#pragma unroll
    for (int i = 0; i < kD / 2; ++i) {
      float2 v = ((const float2*)dwp)[i];
      dw[2 * i] = v.x; dw[2 * i + 1] = v.y;
    }
    float sumu = 0.f, risky = 0.f;
#pragma unroll
    for (int d = 0; d < kD; ++d) {
      float u  = alpha[d];
      float np = u / S[d];
      if (n > 1) dacc += fabsf(np - Np[d]);
      Np[d] = np;
      sumu += u;
      risky += u * (muh + ssq * dw[d]);
    }
    x = x + (x - sumu) * rh + risky;
#pragma unroll
    for (int d = 0; d < kD; ++d) {
      float c = ssq * dw[d];
      S[d]  *= __expf(dp + c);
      SQ[d] *= __expf(dq + c);
    }

    if ((n % kDiff) == 0 && n < kNN) {            // uniform branch: policy update
      const int net = n / kDiff - 1;
      const float* Wi = uW_in + net * kH;
      const float* W2 = uW_h + ((size_t)net * 2 + 0) * kH * kH;
      const float* W3 = uW_h + ((size_t)net * 2 + 1) * kH * kH;
      const float* bL = ub_h + net * 3 * kH;
      const float* gL = bn_g + net * 3 * kH;
      const float* sL = bn_b + net * 3 * kH;
      const float* Wo = uW_out + (size_t)net * kH * kD;
      const float* bo = ub_out + net * kD;

      __syncthreads();
      for (int i = tid; i < 3 * kH; i += kTM) {
        pBias[i]  = bL[i];
        pScale[i] = kInvBN * gL[i];
        pShift[i] = sL[i];
      }
      for (int i = tid; i < kH; i += kTM) pWin[i] = Wi[i];
      for (int i = tid; i < kD; i += kTM) pObias[i] = bo[i];
      // W2 transposed into LDS as WT[n][k], f16
      for (int i = tid; i < kH * kH; i += kTM) {
        int k = i >> 6, nn = i & 63;
        wt[nn * kWS + k] = (_Float16)W2[i];
      }
      __syncthreads();

      // layer 1 (K=1 rank-1): thread-per-row on VALU
#pragma unroll
      for (int j = 0; j < kH; ++j) {
        float t = fmaxf(x * pWin[j] + pBias[j], 0.f);
        actA[tid * kAS + j] = (_Float16)(t * pScale[j] + pShift[j]);
      }
      __syncthreads();

      gemm_hidden(actA, actB, wt, pBias + kH, pScale + kH, pShift + kH, tid);
      __syncthreads();

      for (int i = tid; i < kH * kH; i += kTM) {
        int k = i >> 6, nn = i & 63;
        wt[nn * kWS + k] = (_Float16)W3[i];
      }
      __syncthreads();

      gemm_hidden(actB, actA, wt, pBias + 2 * kH, pScale + 2 * kH, pShift + 2 * kH, tid);
      __syncthreads();

      // output weights transposed; pad rows 20..31 with zeros
      for (int i = tid; i < kH * kD; i += kTM) {
        int k = i / kD, nn = i % kD;
        wtO[nn * kWS + k] = (_Float16)Wo[i];
      }
      for (int i = tid; i < (32 - kD) * kWS; i += kTM)
        wtO[kD * kWS + i] = (_Float16)0.f;
      __syncthreads();

      gemm_out(actA, wtO, pObias, alphaBuf, tid);
      __syncthreads();

#pragma unroll
      for (int d = 0; d < kD; ++d) alpha[d] = alphaBuf[tid * kD + d];
    }
  }

  wsX[b] = x;
  wsD[b] = dacc;
#pragma unroll
  for (int d = 0; d < kD; ++d) {
    wsS[b * kD + d] = S[d];
    wsQ[b * kD + d] = SQ[d];
  }
}

// -------------------- Kernel 2: batch reductions (C0, P0, alpha_tot) --------------------
__global__ __launch_bounds__(256) void reduce_kernel(
    const float* __restrict__ x_in, const float* __restrict__ K_W,
    const float* __restrict__ K_b, const float* __restrict__ opt_W,
    const float* __restrict__ opt_b, const float* __restrict__ wsS,
    const float* __restrict__ wsQ, float* __restrict__ scal) {
  __shared__ float r1[256], r2[256], r3[256];
  const int tid = threadIdx.x;
  const float disc = expf(-kR * 1.0f);
  if ((int)blockIdx.x < kD) {
    const int d = blockIdx.x;
    const float cw = K_W[d], cb = K_b[d], pw = K_W[kD + d], pb = K_b[kD + d];
    float cs = 0.f, ps = 0.f, c0s = 0.f;
    for (int b = tid; b < kB; b += 256) {
      float xi = x_in[b];
      float Kc = 1.f + 0.25f * tanhf(xi * cw + cb);
      float Kp = 1.f + 0.25f * tanhf(xi * pw + pb);
      float sq = wsQ[b * kD + d];
      cs += fmaxf(sq - Kc, 0.f);
      ps += fmaxf(Kp - sq, 0.f);
      if (d == 0) c0s += fmaxf(wsS[b * kD] - Kc, 0.f);
    }
    r1[tid] = cs; r2[tid] = ps; r3[tid] = c0s;
    __syncthreads();
    for (int s = 128; s > 0; s >>= 1) {
      if (tid < s) { r1[tid] += r1[tid + s]; r2[tid] += r2[tid + s]; r3[tid] += r3[tid + s]; }
      __syncthreads();
    }
    if (tid == 0) {
      scal[d]      = disc * r1[0] / (float)kB;           // C0[d]
      scal[kD + d] = disc * r2[0] / (float)kB;           // P0[d]
      if (d == 0) scal[2 * kD] = r3[0] / (float)kB;      // mean(max(S - Kc, 0)) col 0
    }
  } else {
    float s = 0.f;
    for (int b = tid; b < kB; b += 256) {
      float xi = x_in[b];
      float mx = -3.0e38f;
      for (int j = 0; j < 2 * kD + 1; ++j) mx = fmaxf(mx, xi * opt_W[j] + opt_b[j]);
      float sume = 0.f, last = 0.f;
      for (int j = 0; j < 2 * kD + 1; ++j) {
        float e = expf(xi * opt_W[j] + opt_b[j] - mx);
        sume += e;
        if (j == 2 * kD) last = e;
      }
      s += 1.f - last / sume;      // sum of first 2D softmax entries
    }
    r1[tid] = s;
    __syncthreads();
    for (int st = 128; st > 0; st >>= 1) {
      if (tid < st) r1[tid] += r1[tid + st];
      __syncthreads();
    }
    if (tid == 0) {
      float at = r1[0] / (float)kB;
      scal[2 * kD + 1] = at;           // alpha_tot
      scal[2 * kD + 2] = 1.f - at;     // rescale
    }
  }
}

// -------------------- Kernel 3: payoff + terminal wealth, block partials --------------------
__global__ __launch_bounds__(256) void finalize_kernel(
    const float* __restrict__ x_in, const float* __restrict__ K_W,
    const float* __restrict__ K_b, const float* __restrict__ opt_W,
    const float* __restrict__ opt_b, const float* __restrict__ wsX,
    const float* __restrict__ wsD, const float* __restrict__ wsS,
    const float* __restrict__ scal, float* __restrict__ xfin,
    float* __restrict__ part, int nblk) {
  __shared__ float r1[256], r2[256];
  const int tid = threadIdx.x;
  const int b = blockIdx.x * 256 + tid;
  const float xi = x_in[b];
  const float rescale = scal[2 * kD + 2];

  float mx = -3.0e38f;
  for (int j = 0; j < 2 * kD + 1; ++j) mx = fmaxf(mx, xi * opt_W[j] + opt_b[j]);
  float denom = 0.f;
  for (int j = 0; j < 2 * kD + 1; ++j) denom += expf(xi * opt_W[j] + opt_b[j] - mx);

  float payoff = 0.f;
  for (int d = 0; d < kD; ++d) {
    float Sd  = wsS[b * kD + d];
    float Kc  = 1.f + 0.25f * tanhf(xi * K_W[d] + K_b[d]);
    float Kp  = 1.f + 0.25f * tanhf(xi * K_W[kD + d] + K_b[kD + d]);
    float koC = expf(xi * opt_W[d] + opt_b[d] - mx) / denom;
    float koP = expf(xi * opt_W[kD + d] + opt_b[kD + d] - mx) / denom;
    payoff += koC * fmaxf(Sd - Kc, 0.f) / scal[d]
            + koP * fmaxf(Kp - Sd, 0.f) / scal[kD + d];
  }
  float xv = rescale * wsX[b] - kCurt * rescale * wsD[b] + payoff;
  xfin[b] = xv;

  r1[tid] = xv; r2[tid] = xv * xv;
  __syncthreads();
  for (int s = 128; s > 0; s >>= 1) {
    if (tid < s) { r1[tid] += r1[tid + s]; r2[tid] += r2[tid + s]; }
    __syncthreads();
  }
  if (tid == 0) { part[blockIdx.x] = r1[0]; part[nblk + blockIdx.x] = r2[0]; }
}

// -------------------- Kernel 4: sort + percentiles + 7 losses --------------------
__global__ __launch_bounds__(1024) void loss_kernel(
    const float* __restrict__ xfin, const float* __restrict__ part, int npart,
    const float* __restrict__ scal, const float* __restrict__ x_in,
    const float* __restrict__ K_W, const float* __restrict__ K_b,
    float* __restrict__ out) {
  __shared__ float s[kB];                         // 64 KB
  const int tid = threadIdx.x;
  for (int i = tid; i < kB; i += 1024) s[i] = xfin[i];
  __syncthreads();

  // bitonic sort ascending
  for (int k = 2; k <= kB; k <<= 1) {
    for (int j = k >> 1; j > 0; j >>= 1) {
      for (int i = tid; i < kB; i += 1024) {
        int ixj = i ^ j;
        if (ixj > i) {
          float a = s[i], c = s[ixj];
          bool up = ((i & k) == 0);
          if ((a > c) == up) { s[i] = c; s[ixj] = a; }
        }
      }
      __syncthreads();
    }
  }

  if (tid < 32) {
    float ssum = 0.f, ssq = 0.f;
    for (int i = tid; i < npart; i += 32) { ssum += part[i]; ssq += part[npart + i]; }
#pragma unroll
    for (int o = 16; o > 0; o >>= 1) {
      ssum += __shfl_down(ssum, o, 32);
      ssq  += __shfl_down(ssq, o, 32);
    }
    const float q5 = 0.05f * (float)(kB - 1);
    const int i5 = (int)q5;
    const float p5 = s[i5] + (q5 - (float)i5) * (s[i5 + 1] - s[i5]);
    const float q95 = 0.95f * (float)(kB - 1);
    const int i95 = (int)q95;
    const float p95 = s[i95] + (q95 - (float)i95) * (s[i95 + 1] - s[i95]);

    float lo_s = 0.f, hi_s = 0.f; int lo_c = 0, hi_c = 0;
    for (int i = tid; i < kB; i += 32) {
      float v = s[i];
      if (v < p5)  { lo_s += v; ++lo_c; }
      if (v > p95) { hi_s += v; ++hi_c; }
    }
#pragma unroll
    for (int o = 16; o > 0; o >>= 1) {
      lo_s += __shfl_down(lo_s, o, 32);
      hi_s += __shfl_down(hi_s, o, 32);
      lo_c += __shfl_down(lo_c, o, 32);
      hi_c += __shfl_down(hi_c, o, 32);
    }
    if (tid == 0) {
      float mean = ssum / (float)kB;
      out[0] = -mean;                                        // loss1
      out[1] = ssq / (float)kB - mean * mean;                // loss2 (population var)
      out[2] = -lo_s / fmaxf((float)lo_c, 1.f);              // loss3
      out[3] = -hi_s / fmaxf((float)hi_c, 1.f);              // loss4
      out[4] = scal[2 * kD] / scal[0];                       // loss5 = mean(C_T[:,0])
      out[5] = 1.f + 0.25f * tanhf(x_in[0] * K_W[1] + K_b[1]); // loss6 = K[0,1]
      out[6] = scal[2 * kD + 2];                             // loss7 = rescale
    }
  }
}

// -------------------- launch --------------------
extern "C" void kernel_launch(void* const* d_in, const int* in_sizes, int n_in,
                              void* d_out, int out_size, void* d_ws, size_t ws_size,
                              hipStream_t stream) {
  (void)in_sizes; (void)n_in; (void)out_size; (void)ws_size;
  const float* x_in   = (const float*)d_in[0];
  const float* S0     = (const float*)d_in[1];
  const float* dW     = (const float*)d_in[2];
  const float* u0_W   = (const float*)d_in[3];
  const float* u0_b   = (const float*)d_in[4];
  const float* uW_in  = (const float*)d_in[5];
  const float* uW_h   = (const float*)d_in[6];
  const float* ub_h   = (const float*)d_in[7];
  const float* bn_g   = (const float*)d_in[8];
  const float* bn_b   = (const float*)d_in[9];
  const float* uW_out = (const float*)d_in[10];
  const float* ub_out = (const float*)d_in[11];
  const float* opt_W  = (const float*)d_in[12];
  const float* opt_b  = (const float*)d_in[13];
  const float* K_W    = (const float*)d_in[14];
  const float* K_b    = (const float*)d_in[15];

  float* ws   = (float*)d_ws;
  float* wsX  = ws;                       // [B]
  float* wsD  = wsX + kB;                 // [B]
  float* wsS  = wsD + kB;                 // [B*D]
  float* wsQ  = wsS + kB * kD;            // [B*D]
  float* xfin = wsQ + kB * kD;            // [B]
  float* scal = xfin + kB;                // [64]
  float* part = scal + 64;                // [2 * 64]

  sim_kernel<<<kB / kTM, kTM, 0, stream>>>(x_in, S0, dW, u0_W, u0_b, uW_in, uW_h,
                                           ub_h, bn_g, bn_b, uW_out, ub_out,
                                           wsX, wsD, wsS, wsQ);
  reduce_kernel<<<kD + 1, 256, 0, stream>>>(x_in, K_W, K_b, opt_W, opt_b,
                                            wsS, wsQ, scal);
  const int nblk = kB / 256;
  finalize_kernel<<<nblk, 256, 0, stream>>>(x_in, K_W, K_b, opt_W, opt_b,
                                            wsX, wsD, wsS, scal, xfin, part, nblk);
  loss_kernel<<<1, 1024, 0, stream>>>(xfin, part, nblk, scal, x_in, K_W, K_b,
                                      (float*)d_out);
}
